// SparseAttention4D_2937757630658
// MI455X (gfx1250) — matrix operand order repaired and checked by reading, not compiled, not run on hardware
//
#include <hip/hip_runtime.h>

typedef __attribute__((ext_vector_type(16))) _Float16 v16h;
typedef __attribute__((ext_vector_type(8)))  _Float16 v8h;
typedef __attribute__((ext_vector_type(8)))  float    v8f;

#define B_ 4
#define C_ 8
#define S_ 512
#define E_ 512
#define H_ 8
#define HD_ 64
#define NROWS_ (B_ * C_ * S_)   // 16384

static __device__ __forceinline__ v8f wmma_f16(v16h a, v16h b, v8f c) {
  // D = A(16x32 f16) * B(32x16 f16) + C(16x16 f32)
  return __builtin_amdgcn_wmma_f32_16x16x32_f16(false, a, false, b, (short)0, c,
                                                false, false);
}

// CDNA5 async global->LDS copy (16B per lane), tracked by ASYNCcnt.
static __device__ __forceinline__ void async_ld_b128(unsigned ldsoff,
                                                     const void* gptr) {
  unsigned long long ga = (unsigned long long)(size_t)gptr;
  asm volatile("global_load_async_to_lds_b128 %0, %1, off"
               :: "v"(ldsoff), "v"(ga) : "memory");
}
static __device__ __forceinline__ void wait_async0() {
  asm volatile("s_wait_asynccnt 0x0" ::: "memory");
}
#define LDSOFF(p) ((unsigned)(size_t)(p))

// ---------------------------------------------------------------------------
// GEMM: Out[n, e] = (OutT)(sum_k X[n,k] * W[e,k] + bias[e]).
// X: [16384 x 512] InT row-major, W/bias f32. TRANS_OUT stores Out column-major
// ([E_][NROWS_]) so downstream B-fragments become memory-contiguous.
// Block = 256 thr (8 wave32), tile 128x64, one 16-row strip per wave.
// ---------------------------------------------------------------------------
template <typename InT, typename OutT, bool TRANS_OUT>
__global__ __launch_bounds__(256)
void gemm_xwt_bias(const InT* __restrict__ X, const float* __restrict__ W,
                   const float* __restrict__ bias, OutT* __restrict__ Out) {
  __shared__ alignas(32) _Float16 Alds[8][32][16];  // 8 wave strips, A-frag order
  __shared__ alignas(32) _Float16 Blds[4][32][16];  // 4 n-tiles, B-frag order

  const int t = threadIdx.x;
  const int w = t >> 5;
  const int lane = t & 31;
  const int hf = lane >> 4;
  const int ln = lane & 15;
  const int m0 = blockIdx.x * 128;
  const int n0 = blockIdx.y * 64;

  v8f acc[4] = {v8f{}, v8f{}, v8f{}, v8f{}};

  const int arow = t >> 1;          // 0..127
  const int akb  = (t & 1) * 16;    // 0 or 16
  const int bn   = t >> 2;          // 0..63
  const int bkb  = (t & 3) * 8;     // 0,8,16,24

  for (int kc = 0; kc < E_; kc += 32) {
    const InT* xr = X + (size_t)(m0 + arow) * E_ + kc + akb;
    const float* wr = W + (size_t)(n0 + bn) * E_ + kc + bkb;
    if (kc + 32 < E_) {             // prefetch next k-tile -> global_prefetch_b8
      __builtin_prefetch(xr + 32, 0, 0);
      __builtin_prefetch(wr + 32, 0, 0);
    }
    #pragma unroll
    for (int q = 0; q < 16; ++q) {  // A tile 128x32, fragment-order scatter
      int k = akb + q;
      Alds[arow >> 4][(arow & 15) + 16 * ((k >> 3) & 1)]
          [(k & 7) + 8 * (k >> 4)] = (_Float16)xr[q];
    }
    #pragma unroll
    for (int q = 0; q < 8; ++q) {   // B tile 32x64: B(k,n) = W[n0+n][kc+k]
      int k = bkb + q;
      Blds[bn >> 4][(bn & 15) + 16 * ((k >> 3) & 1)][(k & 7) + 8 * (k >> 4)] = (_Float16)wr[q];
    }
    __syncthreads();
    v16h afr = *(const v16h*)&Alds[w][lane][0];
    #pragma unroll
    for (int nt = 0; nt < 4; ++nt) {
      v16h bfr = *(const v16h*)&Blds[nt][lane][0];
      acc[nt] = wmma_f16(afr, bfr, acc[nt]);
    }
    __syncthreads();
  }

  // Epilogue: D element (r, lane) -> row m0+16w+8*hf+r, col n0+16nt+ln.
  #pragma unroll
  for (int nt = 0; nt < 4; ++nt) {
    int col = n0 + nt * 16 + ln;
    float bb = bias[col];
    int rb = m0 + w * 16 + 8 * hf;
    #pragma unroll
    for (int r = 0; r < 8; ++r) {
      float v = acc[nt][r] + bb;
      if (TRANS_OUT) Out[(size_t)col * NROWS_ + rb + r] = (OutT)v;
      else           Out[(size_t)(rb + r) * E_ + col]   = (OutT)v;
    }
  }
}

// ---------------------------------------------------------------------------
// Sparse attention. Block = (b,h,c, 128-row i-chunk); 8 waves, one 16-row
// i-tile each, dense j loop in lockstep over 32-row K/V chunks. Tiles staged
// global->LDS in WMMA fragment order with async b128 copies (wave 0 issues,
// ASYNCcnt wait + barrier). qh/kh: f16 row-major; vhT: f16 [512 x 16384].
// ---------------------------------------------------------------------------
__global__ __launch_bounds__(256)
void sparse_attn(const _Float16* __restrict__ qh, const _Float16* __restrict__ kh,
                 const _Float16* __restrict__ vhT, float* __restrict__ attn,
                 _Float16* __restrict__ ctxh) {
  __shared__ alignas(32) _Float16 Kc[2][2][32][16];   // K chunk (2 j-tiles, 2 d-halves)
  __shared__ alignas(32) _Float16 Vf[4][32][16];      // V chunk (4 n-tiles)
  __shared__ alignas(32) _Float16 Ps[8][32][16];      // per-wave P chunk (A-frag)

  const int t = threadIdx.x, w = t >> 5, lane = t & 31;
  const int hf = lane >> 4, ln = lane & 15;
  const int bhc = blockIdx.x;                 // ((b*H + h)*C + c)
  const int b = bhc / (H_ * C_);
  const int h = (bhc / C_) % H_;
  const int c = bhc % C_;
  const size_t rowbase = (size_t)(b * C_ + c) * S_;
  const int hcol = h * HD_;
  const int i0 = (blockIdx.y * 8 + w) * 16;

  // --- Q A-fragments: per lane two contiguous 16B runs per fragment ---
  v16h aq0, aq1;
  {
    const _Float16* qrow = qh + (rowbase + i0 + ln) * E_ + hcol + 16 * hf;
    v8h lo0 = *(const v8h*)(qrow);
    v8h hi0 = *(const v8h*)(qrow + 8);
    v8h lo1 = *(const v8h*)(qrow + 32);
    v8h hi1 = *(const v8h*)(qrow + 40);
    aq0 = __builtin_shufflevector(lo0, hi0, 0,1,2,3,4,5,6,7,8,9,10,11,12,13,14,15);
    aq1 = __builtin_shufflevector(lo1, hi1, 0,1,2,3,4,5,6,7,8,9,10,11,12,13,14,15);
  }

  // --- Phase 1: scores, branch-free online per-lane max / sum-of-exp ---
  float mx[8], sm[8];
  #pragma unroll
  for (int r = 0; r < 8; ++r) { mx[r] = -1.0e30f; sm[r] = 0.0f; }

  for (int jc = 0; jc < 16; ++jc) {
    const int j0c = jc * 32;
    if (w == 0) {
      #pragma unroll
      for (int jt2 = 0; jt2 < 2; ++jt2) {
        const _Float16* kbase =
            kh + (rowbase + j0c + jt2 * 16 + ln) * (size_t)E_ + hcol + hf * 16;
        #pragma unroll
        for (int fd = 0; fd < 2; ++fd) {
          unsigned l = LDSOFF(&Kc[jt2][fd][lane][0]);
          async_ld_b128(l,      kbase + fd * 32);
          async_ld_b128(l + 16, kbase + fd * 32 + 8);
        }
      }
      wait_async0();
    }
    __syncthreads();
    #pragma unroll
    for (int jt2 = 0; jt2 < 2; ++jt2) {
      v8f acc = {};
      acc = wmma_f16(aq0, *(const v16h*)&Kc[jt2][0][lane][0], acc);
      acc = wmma_f16(aq1, *(const v16h*)&Kc[jt2][1][lane][0], acc);
      int j = j0c + jt2 * 16 + ln;
      #pragma unroll
      for (int r = 0; r < 8; ++r) {
        int i = i0 + 8 * hf + r;
        float s = acc[r] * 0.125f;   // 1/sqrt(64)
        bool allowed = ((j <= i) && (j + 32 >= i)) ||
                       (((j & 3) == 0) && (j < i));
        s = allowed ? s : -1.0e30f;
        float nm = fmaxf(mx[r], s);                       // branch-free update
        sm[r] = sm[r] * __expf(mx[r] - nm) + __expf(s - nm);
        mx[r] = nm;
      }
    }
    __syncthreads();
  }

  // --- cross-lane (16-lane column group) max/sum merge ---
  #pragma unroll
  for (int r = 0; r < 8; ++r) {
    float m = mx[r], s = sm[r];
    #pragma unroll
    for (int off = 1; off <= 8; off <<= 1) {
      float om = __shfl_xor(m, off, 32);
      float os = __shfl_xor(s, off, 32);
      float nm = fmaxf(m, om);
      s = s * __expf(m - nm) + os * __expf(om - nm);
      m = nm;
    }
    mx[r] = m;
    sm[r] = 1.0f / s;   // reciprocal row sum
  }

  // --- Phase 2: recompute scores, emit attn, accumulate context = P * V ---
  v8f cacc[4] = {v8f{}, v8f{}, v8f{}, v8f{}};
  for (int jc = 0; jc < 16; ++jc) {
    const int j0c = jc * 32;
    if (w == 0) {
      #pragma unroll
      for (int jt2 = 0; jt2 < 2; ++jt2) {
        const _Float16* kbase =
            kh + (rowbase + j0c + jt2 * 16 + ln) * (size_t)E_ + hcol + hf * 16;
        #pragma unroll
        for (int fd = 0; fd < 2; ++fd) {
          unsigned l = LDSOFF(&Kc[jt2][fd][lane][0]);
          async_ld_b128(l,      kbase + fd * 32);
          async_ld_b128(l + 16, kbase + fd * 32 + 8);
        }
      }
      #pragma unroll
      for (int nt = 0; nt < 4; ++nt) {
        // V B-frag slice: vhT[d = hcol+nt*16+ln][rows j0c + hf*16 + [0..15]]
        const _Float16* vbase =
            vhT + (size_t)(hcol + nt * 16 + ln) * NROWS_ + rowbase + j0c + hf * 16;
        unsigned l = LDSOFF(&Vf[nt][lane][0]);
        async_ld_b128(l,      vbase);
        async_ld_b128(l + 16, vbase + 8);
      }
      wait_async0();
    }
    __syncthreads();
    #pragma unroll
    for (int jt2 = 0; jt2 < 2; ++jt2) {
      v8f acc = {};
      acc = wmma_f16(aq0, *(const v16h*)&Kc[jt2][0][lane][0], acc);
      acc = wmma_f16(aq1, *(const v16h*)&Kc[jt2][1][lane][0], acc);
      int j = j0c + jt2 * 16 + ln;
      #pragma unroll
      for (int r = 0; r < 8; ++r) {
        int i = i0 + 8 * hf + r;
        float s = acc[r] * 0.125f;
        bool allowed = ((j <= i) && (j + 32 >= i)) ||
                       (((j & 3) == 0) && (j < i));
        s = allowed ? s : -1.0e30f;
        float p = __expf(s - mx[r]) * sm[r];
        attn[((size_t)bhc * S_ + i) * S_ + j] = p;
        int m  = 8 * hf + r;
        int kk = jt2 * 16 + ln;
        Ps[w][m + 16 * (kk >> 4)][kk & 15] = (_Float16)p;
      }
    }
    __syncthreads();
    v16h pf = *(const v16h*)&Ps[w][lane][0];
    #pragma unroll
    for (int nt = 0; nt < 4; ++nt)
      cacc[nt] = wmma_f16(pf, *(const v16h*)&Vf[nt][lane][0], cacc[nt]);
    __syncthreads();
  }

  // --- context store (f16): ctxh[b,c,s, h*64+d] ---
  #pragma unroll
  for (int nt = 0; nt < 4; ++nt) {
    int d = nt * 16 + ln;
    #pragma unroll
    for (int r = 0; r < 8; ++r) {
      int i = i0 + 8 * hf + r;
      ctxh[(rowbase + i) * E_ + hcol + d] = (_Float16)cacc[nt][r];
    }
  }
}

// ---------------------------------------------------------------------------
extern "C" void kernel_launch(void* const* d_in, const int* in_sizes, int n_in,
                              void* d_out, int out_size, void* d_ws, size_t ws_size,
                              hipStream_t stream) {
  (void)in_sizes; (void)n_in; (void)out_size; (void)ws_size;
  const float* query = (const float*)d_in[0];
  const float* key_  = (const float*)d_in[1];
  const float* value = (const float*)d_in[2];
  const float* Wq = (const float*)d_in[3];  const float* bq = (const float*)d_in[4];
  const float* Wk = (const float*)d_in[5];  const float* bk = (const float*)d_in[6];
  const float* Wv = (const float*)d_in[7];  const float* bv = (const float*)d_in[8];
  const float* Wo = (const float*)d_in[9];  const float* bo = (const float*)d_in[10];

  float* out  = (float*)d_out;                          // [B,C,S,E]
  float* attn = out + (size_t)B_ * C_ * S_ * E_;        // [B,H,C,S,S]

  const size_t NE = (size_t)NROWS_ * E_;                // 8,388,608 elements
  _Float16* wsh = (_Float16*)d_ws;
  _Float16* qh   = wsh;            // f16 [16384 x 512]
  _Float16* kh   = wsh + NE;       // f16 [16384 x 512]
  _Float16* vhT  = wsh + 2 * NE;   // f16 [512 x 16384] (transposed)
  _Float16* ctxh = wsh + 3 * NE;   // f16 [16384 x 512]

  dim3 gblk(256);
  dim3 ggrid(NROWS_ / 128, E_ / 64);
  gemm_xwt_bias<float, _Float16, false><<<ggrid, gblk, 0, stream>>>(query, Wq, bq, qh);
  gemm_xwt_bias<float, _Float16, false><<<ggrid, gblk, 0, stream>>>(key_,  Wk, bk, kh);
  gemm_xwt_bias<float, _Float16, true ><<<ggrid, gblk, 0, stream>>>(value, Wv, bv, vhT);

  dim3 agrid(B_ * H_ * C_, S_ / 128);
  sparse_attn<<<agrid, dim3(256), 0, stream>>>(qh, kh, vhT, attn, ctxh);

  gemm_xwt_bias<_Float16, float, false><<<ggrid, gblk, 0, stream>>>(ctxh, Wo, bo, out);
}
